// CustomTransformerEncoderLayer_65549790872156
// MI455X (gfx1250) — matrix operand order repaired and checked by reading, not compiled, not run on hardware
//
#include <hip/hip_runtime.h>
#include <cmath>

// ---------- types ----------
typedef __attribute__((ext_vector_type(16))) __bf16 v16bf;
typedef __attribute__((ext_vector_type(8)))  __bf16 v8bf;
typedef __attribute__((ext_vector_type(8)))  float  v8f;

#define TILE 128
#define LDSS 40   // padded LDS row stride in bf16 elements (80B -> conflict-free, 16B aligned)

// CDNA5 async global->LDS path (ASYNCcnt), if the toolchain exposes it.
#if defined(__has_builtin)
#if __has_builtin(__builtin_amdgcn_global_load_async_to_lds_b128) && \
    __has_builtin(__builtin_amdgcn_s_wait_asynccnt)
#define USE_ASYNC_LDS 1
#endif
#endif

#ifdef USE_ASYNC_LDS
typedef int v4i_g __attribute__((vector_size(16)));   // generic-vector int4, matches builtin
#define AS1P(p) ((__attribute__((address_space(1))) v4i_g*)(p))
#define AS3P(p) ((__attribute__((address_space(3))) v4i_g*)(p))
#endif

__device__ __forceinline__ v16bf frag16(const ushort* p, int o0, int o1) {
    v8bf lo = *reinterpret_cast<const v8bf*>(p + o0);
    v8bf hi = *reinterpret_cast<const v8bf*>(p + o1);
    return __builtin_shufflevector(lo, hi, 0,1,2,3,4,5,6,7,8,9,10,11,12,13,14,15);
}

// ---------------------------------------------------------------------------
// Generic 128x128 block GEMM: C[M,N] = A[M,K](bf16,row-major) * Bt[N,K](bf16)^T
// 8 waves, each wave computes a 32x64 tile via 2x4 v_wmma_f32_16x16x32_bf16.
// Double-buffered LDS; next K-tile streams in via global_load_async_to_lds_b128
// (ASYNCcnt) while WMMA consumes the current tile.
// MODE epilogues:
//   0: out_bf16 = acc + bias                       (Q, K projections)
//   1: out_bf16 transposed per-batch = acc + bias  (V projection -> vT[b][d][s])
//   2: out_f32  = acc * scale                      (attention scores)
//   3: out_f32  = acc + addsrc                     (attn@V + residual)
//   4: out_bf16 = gelu(acc + bias)                 (FFN1)
//   5: out_f32  = acc + bias + addsrc              (FFN2 + residual)
// ---------------------------------------------------------------------------
template<int MODE>
__global__ __launch_bounds__(256) void gemm_bf16_wmma(
    const __bf16* __restrict__ A, const __bf16* __restrict__ Bt,
    int M, int N, int K,
    long long sA, long long sB, long long sO, long long sAdd,
    const float* __restrict__ bias, const float* __restrict__ addsrc,
    float scale, float* __restrict__ outF, __bf16* __restrict__ outB, int Sdim)
{
    __shared__ __align__(16) ushort As[2][TILE * LDSS];
    __shared__ __align__(16) ushort Bs[2][TILE * LDSS];

    const long long z = blockIdx.z;
    A  += z * sA;
    Bt += z * sB;
    if (outF)   outF   += z * sO;
    if (addsrc) addsrc += z * sAdd;

    const int tid  = threadIdx.x;
    const int lane = tid & 31;
    const int wid  = tid >> 5;
    const int wm   = wid & 3;    // 4 waves along M  (32 rows each)
    const int wn   = wid >> 2;   // 2 waves along N  (64 cols each)
    const int half = lane >> 4;
    const int l16  = lane & 15;

    const int row0 = blockIdx.y * TILE;
    const int col0 = blockIdx.x * TILE;

    v8f acc[2][4];
    const v8f zacc = {0.f,0.f,0.f,0.f,0.f,0.f,0.f,0.f};
    #pragma unroll
    for (int mt = 0; mt < 2; ++mt)
        #pragma unroll
        for (int nt = 0; nt < 4; ++nt)
            acc[mt][nt] = zacc;

    const __bf16* gA = A  + (size_t)row0 * K;
    const __bf16* gB = Bt + (size_t)col0 * K;

    // Each thread stages 4x16B per K-tile: rows r and r+64 of both operands.
    const int    r   = tid >> 2;
    const int    cc  = (tid & 3) * 8;
    const size_t o0  = (size_t)r * K + cc;
    const size_t o1  = (size_t)(r + 64) * K + cc;
    const int    l0  = r * LDSS + cc;
    const int    l1  = (r + 64) * LDSS + cc;

    // ---- prologue: stage K-tile 0 into buffer 0
#ifdef USE_ASYNC_LDS
    __builtin_amdgcn_global_load_async_to_lds_b128(AS1P(gA + o0), AS3P(&As[0][l0]), 0, 0);
    __builtin_amdgcn_global_load_async_to_lds_b128(AS1P(gA + o1), AS3P(&As[0][l1]), 0, 0);
    __builtin_amdgcn_global_load_async_to_lds_b128(AS1P(gB + o0), AS3P(&Bs[0][l0]), 0, 0);
    __builtin_amdgcn_global_load_async_to_lds_b128(AS1P(gB + o1), AS3P(&Bs[0][l1]), 0, 0);
#else
    {
        const uint4 a0 = *reinterpret_cast<const uint4*>(gA + o0);
        const uint4 a1 = *reinterpret_cast<const uint4*>(gA + o1);
        const uint4 b0 = *reinterpret_cast<const uint4*>(gB + o0);
        const uint4 b1 = *reinterpret_cast<const uint4*>(gB + o1);
        *reinterpret_cast<uint4*>(&As[0][l0]) = a0;
        *reinterpret_cast<uint4*>(&As[0][l1]) = a1;
        *reinterpret_cast<uint4*>(&Bs[0][l0]) = b0;
        *reinterpret_cast<uint4*>(&Bs[0][l1]) = b1;
    }
#endif

    int p = 0;
    for (int k0 = 0; k0 < K; k0 += 32, p ^= 1) {
#ifdef USE_ASYNC_LDS
        __builtin_amdgcn_s_wait_asynccnt(0);   // my async writes to buffer p landed
#endif
        __syncthreads();                        // everyone's writes landed
        const bool more = (k0 + 32) < K;

#ifdef USE_ASYNC_LDS
        if (more) {                             // stream next tile into buffer p^1
            const int kn = k0 + 32;
            __builtin_amdgcn_global_load_async_to_lds_b128(AS1P(gA + o0 + kn), AS3P(&As[p ^ 1][l0]), 0, 0);
            __builtin_amdgcn_global_load_async_to_lds_b128(AS1P(gA + o1 + kn), AS3P(&As[p ^ 1][l1]), 0, 0);
            __builtin_amdgcn_global_load_async_to_lds_b128(AS1P(gB + o0 + kn), AS3P(&Bs[p ^ 1][l0]), 0, 0);
            __builtin_amdgcn_global_load_async_to_lds_b128(AS1P(gB + o1 + kn), AS3P(&Bs[p ^ 1][l1]), 0, 0);
        }
#else
        uint4 a0, a1, b0, b1;                   // issue loads now, store after WMMA
        if (more) {
            const int kn = k0 + 32;
            a0 = *reinterpret_cast<const uint4*>(gA + o0 + kn);
            a1 = *reinterpret_cast<const uint4*>(gA + o1 + kn);
            b0 = *reinterpret_cast<const uint4*>(gB + o0 + kn);
            b1 = *reinterpret_cast<const uint4*>(gB + o1 + kn);
        }
#endif
        if (k0 + 64 < K) {                      // warm L2 two tiles ahead
            __builtin_prefetch((const void*)(gA + o0 + k0 + 64), 0, 1);
            __builtin_prefetch((const void*)(gB + o0 + k0 + 64), 0, 1);
        }

        // ---- fragments per ISA 16-bit A 16x32 / B 32x16 layouts, from buffer p
        v16bf af[2], bfg[4];
        #pragma unroll
        for (int mt = 0; mt < 2; ++mt) {
            const ushort* pa = &As[p][(wm * 32 + mt * 16 + l16) * LDSS];
            af[mt] = frag16(pa, half * 8, 16 + half * 8);
        }
        #pragma unroll
        for (int nt = 0; nt < 4; ++nt) {
            const ushort* pb = &Bs[p][(wn * 64 + nt * 16 + l16) * LDSS];
            bfg[nt] = frag16(pb, half * 8, 16 + half * 8);
        }
        #pragma unroll
        for (int mt = 0; mt < 2; ++mt)
            #pragma unroll
            for (int nt = 0; nt < 4; ++nt)
                acc[mt][nt] = __builtin_amdgcn_wmma_f32_16x16x32_bf16(
                    false, af[mt], false, bfg[nt], (short)0, acc[mt][nt], false, false);

#ifndef USE_ASYNC_LDS
        if (more) {                             // global latency hidden behind WMMA above
            *reinterpret_cast<uint4*>(&As[p ^ 1][l0]) = a0;
            *reinterpret_cast<uint4*>(&As[p ^ 1][l1]) = a1;
            *reinterpret_cast<uint4*>(&Bs[p ^ 1][l0]) = b0;
            *reinterpret_cast<uint4*>(&Bs[p ^ 1][l1]) = b1;
        }
#endif
    }

    // ---- epilogue: C layout -> lane holds (m = half*8 + r, n = l16) per VGPR r
    #pragma unroll
    for (int mt = 0; mt < 2; ++mt) {
        #pragma unroll
        for (int nt = 0; nt < 4; ++nt) {
            #pragma unroll
            for (int rr = 0; rr < 8; ++rr) {
                const int m = row0 + wm * 32 + mt * 16 + half * 8 + rr;
                const int n = col0 + wn * 64 + nt * 16 + l16;
                const float v = acc[mt][nt][rr];
                if constexpr (MODE == 0) {
                    outB[(size_t)m * N + n] = (__bf16)(v + bias[n]);
                } else if constexpr (MODE == 1) {
                    const int bb = m / Sdim, s = m % Sdim;
                    outB[((size_t)bb * N + n) * Sdim + s] = (__bf16)(v + bias[n]);
                } else if constexpr (MODE == 2) {
                    outF[(size_t)m * N + n] = v * scale;
                } else if constexpr (MODE == 3) {
                    const size_t idx = (size_t)m * N + n;
                    outF[idx] = v + addsrc[idx];
                } else if constexpr (MODE == 4) {
                    const float h = v + bias[n];
                    outB[(size_t)m * N + n] =
                        (__bf16)(0.5f * h * (1.0f + erff(h * 0.7071067811865475f)));
                } else {  // MODE 5
                    const size_t idx = (size_t)m * N + n;
                    outF[idx] = v + bias[n] + addsrc[idx];
                }
            }
        }
    }
}

// ---------------------------------------------------------------------------
// softmax(scores + gumbel) per row of S=1024; writes bf16 attn weights.
// ---------------------------------------------------------------------------
__global__ __launch_bounds__(256) void softmax_gumbel_kernel(
    const float* __restrict__ scores, const float* __restrict__ gu,
    __bf16* __restrict__ attn, int S)
{
    const int row = blockIdx.x;
    const int tid = threadIdx.x;
    __shared__ float red[256];

    const float* sp = scores + (size_t)row * S;
    const float* gp = gu     + (size_t)row * S;

    float v[4];
    float m = -3.402823466e38f;
    #pragma unroll
    for (int j = 0; j < 4; ++j) {
        const int i = tid + j * 256;
        const float g = -__logf(-__logf(gp[i]));
        v[j] = sp[i] + g;
        m = fmaxf(m, v[j]);
    }
    red[tid] = m; __syncthreads();
    for (int s = 128; s > 0; s >>= 1) {
        if (tid < s) red[tid] = fmaxf(red[tid], red[tid + s]);
        __syncthreads();
    }
    m = red[0]; __syncthreads();

    float e[4], sum = 0.f;
    #pragma unroll
    for (int j = 0; j < 4; ++j) { e[j] = __expf(v[j] - m); sum += e[j]; }
    red[tid] = sum; __syncthreads();
    for (int s = 128; s > 0; s >>= 1) {
        if (tid < s) red[tid] += red[tid + s];
        __syncthreads();
    }
    const float inv = 1.0f / red[0];

    __bf16* op = attn + (size_t)row * S;
    #pragma unroll
    for (int j = 0; j < 4; ++j) op[tid + j * 256] = (__bf16)(e[j] * inv);
}

// ---------------------------------------------------------------------------
// LayerNorm over D=768 per row; optionally also emits bf16 copy for next GEMM.
// ---------------------------------------------------------------------------
template<int WRITE_BF>
__global__ __launch_bounds__(256) void layernorm_kernel(
    const float* __restrict__ in, const float* __restrict__ w, const float* __restrict__ b,
    float* __restrict__ outF, __bf16* __restrict__ outB, int D)
{
    const int row = blockIdx.x;
    const int tid = threadIdx.x;
    __shared__ float red[256];
    const float* x = in + (size_t)row * D;

    float lv[3], s = 0.f;
    #pragma unroll
    for (int j = 0; j < 3; ++j) { lv[j] = x[tid + j * 256]; s += lv[j]; }
    red[tid] = s; __syncthreads();
    for (int t = 128; t > 0; t >>= 1) { if (tid < t) red[tid] += red[tid + t]; __syncthreads(); }
    const float mean = red[0] * (1.0f / 768.0f);
    __syncthreads();

    float sq = 0.f;
    #pragma unroll
    for (int j = 0; j < 3; ++j) { const float d = lv[j] - mean; sq += d * d; }
    red[tid] = sq; __syncthreads();
    for (int t = 128; t > 0; t >>= 1) { if (tid < t) red[tid] += red[tid + t]; __syncthreads(); }
    const float rstd = rsqrtf(red[0] * (1.0f / 768.0f) + 1e-5f);

    #pragma unroll
    for (int j = 0; j < 3; ++j) {
        const int i = tid + j * 256;
        const float o = (lv[j] - mean) * rstd * w[i] + b[i];
        outF[(size_t)row * D + i] = o;
        if constexpr (WRITE_BF) outB[(size_t)row * D + i] = (__bf16)o;
    }
}

// ---------- casts ----------
__global__ __launch_bounds__(256) void cast_f32_bf16(
    const float* __restrict__ in, __bf16* __restrict__ out, long long n)
{
    const long long i = (long long)blockIdx.x * 256 + threadIdx.x;
    if (i < n) out[i] = (__bf16)in[i];
}

__global__ __launch_bounds__(256) void cast_transpose_f32_bf16(
    const float* __restrict__ in, __bf16* __restrict__ out, int K, int N)
{
    const long long i = (long long)blockIdx.x * 256 + threadIdx.x;
    if (i < (long long)K * N) {
        const int k = (int)(i / N), n = (int)(i % N);
        out[(size_t)n * K + k] = (__bf16)in[i];
    }
}

// ---------------------------------------------------------------------------
extern "C" void kernel_launch(void* const* d_in, const int* in_sizes, int n_in,
                              void* d_out, int out_size, void* d_ws, size_t ws_size,
                              hipStream_t stream) {
    (void)in_sizes; (void)n_in; (void)out_size; (void)ws_size;
    constexpr int Bn = 16, S = 1024, Dm = 768, DFF = 3072;
    const size_t MS = (size_t)Bn * S;   // 16384 rows

    const float* src  = (const float*)d_in[0];
    const float* gu   = (const float*)d_in[1];
    const float* wq   = (const float*)d_in[2];
    const float* bq   = (const float*)d_in[3];
    const float* wk   = (const float*)d_in[4];
    const float* bk   = (const float*)d_in[5];
    const float* wv   = (const float*)d_in[6];
    const float* bv   = (const float*)d_in[7];
    const float* w1   = (const float*)d_in[8];
    const float* b1   = (const float*)d_in[9];
    const float* w2   = (const float*)d_in[10];
    const float* b2   = (const float*)d_in[11];
    const float* ln1w = (const float*)d_in[12];
    const float* ln1b = (const float*)d_in[13];
    const float* ln2w = (const float*)d_in[14];
    const float* ln2b = (const float*)d_in[15];
    float* out = (float*)d_out;

    char* ws = (char*)d_ws;
    size_t off = 0;
    auto alloc = [&](size_t bytes) -> void* {
        void* p = (void*)(ws + off);
        off += (bytes + 255) & ~(size_t)255;
        return p;
    };
    __bf16* srcb  = (__bf16*)alloc(MS * Dm * 2);
    __bf16* wqT   = (__bf16*)alloc((size_t)Dm * Dm * 2);
    __bf16* wkT   = (__bf16*)alloc((size_t)Dm * Dm * 2);
    __bf16* wvT   = (__bf16*)alloc((size_t)Dm * Dm * 2);
    __bf16* w1T   = (__bf16*)alloc((size_t)Dm * DFF * 2);   // [DFF][Dm]
    __bf16* w2T   = (__bf16*)alloc((size_t)Dm * DFF * 2);   // [Dm][DFF]
    __bf16* qb    = (__bf16*)alloc(MS * Dm * 2);
    __bf16* kb    = (__bf16*)alloc(MS * Dm * 2);
    __bf16* vT    = (__bf16*)alloc(MS * Dm * 2);            // [B][Dm][S]
    float*  sc    = (float*) alloc((size_t)Bn * S * S * 4);
    __bf16* attnb = (__bf16*)alloc((size_t)Bn * S * S * 2);
    float*  y     = (float*) alloc(MS * Dm * 4);            // reused as z for FFN2
    float*  x     = (float*) alloc(MS * Dm * 4);
    __bf16* xb    = (__bf16*)alloc(MS * Dm * 2);
    __bf16* hb    = (__bf16*)alloc(MS * DFF * 2);

    const dim3 blk(256);

    // ---- prep: bf16 casts + weight transposes
    {
        long long n = (long long)MS * Dm;
        cast_f32_bf16<<<dim3((unsigned)((n + 255) / 256)), blk, 0, stream>>>(src, srcb, n);
    }
    {
        long long nd = (long long)Dm * Dm;
        dim3 g((unsigned)((nd + 255) / 256));
        cast_transpose_f32_bf16<<<g, blk, 0, stream>>>(wq, wqT, Dm, Dm);
        cast_transpose_f32_bf16<<<g, blk, 0, stream>>>(wk, wkT, Dm, Dm);
        cast_transpose_f32_bf16<<<g, blk, 0, stream>>>(wv, wvT, Dm, Dm);
        long long nf = (long long)Dm * DFF;
        dim3 gf((unsigned)((nf + 255) / 256));
        cast_transpose_f32_bf16<<<gf, blk, 0, stream>>>(w1, w1T, Dm, DFF);   // -> [DFF][Dm]
        cast_transpose_f32_bf16<<<gf, blk, 0, stream>>>(w2, w2T, DFF, Dm);   // -> [Dm][DFF]
    }

    // ---- Q, K projections (bias -> bf16)
    gemm_bf16_wmma<0><<<dim3(Dm / 128, (unsigned)(MS / 128), 1), blk, 0, stream>>>(
        srcb, wqT, (int)MS, Dm, Dm, 0, 0, 0, 0, bq, nullptr, 1.0f, nullptr, qb, S);
    gemm_bf16_wmma<0><<<dim3(Dm / 128, (unsigned)(MS / 128), 1), blk, 0, stream>>>(
        srcb, wkT, (int)MS, Dm, Dm, 0, 0, 0, 0, bk, nullptr, 1.0f, nullptr, kb, S);
    // ---- V projection, stored transposed per batch: vT[b][d][s]
    gemm_bf16_wmma<1><<<dim3(Dm / 128, (unsigned)(MS / 128), 1), blk, 0, stream>>>(
        srcb, wvT, (int)MS, Dm, Dm, 0, 0, 0, 0, bv, nullptr, 1.0f, nullptr, vT, S);

    // ---- scores = (Q @ K^T) / sqrt(D), batched over B (K already Bt-layout [S][D])
    const float scale = 1.0f / sqrtf((float)Dm);
    gemm_bf16_wmma<2><<<dim3(S / 128, S / 128, Bn), blk, 0, stream>>>(
        qb, kb, S, S, Dm,
        (long long)S * Dm, (long long)S * Dm, (long long)S * S, 0,
        nullptr, nullptr, scale, sc, nullptr, S);

    // ---- gumbel softmax -> bf16 attention weights
    softmax_gumbel_kernel<<<dim3((unsigned)(Bn * S)), blk, 0, stream>>>(sc, gu, attnb, S);

    // ---- attn @ V + residual(src) -> y (fp32), batched
    gemm_bf16_wmma<3><<<dim3(Dm / 128, S / 128, Bn), blk, 0, stream>>>(
        attnb, vT, S, Dm, S,
        (long long)S * S, (long long)Dm * S, (long long)S * Dm, (long long)S * Dm,
        nullptr, src, 1.0f, y, nullptr, S);

    // ---- LN1 -> x (fp32) + xb (bf16)
    layernorm_kernel<1><<<dim3((unsigned)MS), blk, 0, stream>>>(y, ln1w, ln1b, x, xb, Dm);

    // ---- FFN1: gelu(x @ w1 + b1) -> hb (bf16)
    gemm_bf16_wmma<4><<<dim3(DFF / 128, (unsigned)(MS / 128), 1), blk, 0, stream>>>(
        xb, w1T, (int)MS, DFF, Dm, 0, 0, 0, 0, b1, nullptr, 1.0f, nullptr, hb, S);

    // ---- FFN2: h @ w2 + b2 + x -> y (reused as z)
    gemm_bf16_wmma<5><<<dim3(Dm / 128, (unsigned)(MS / 128), 1), blk, 0, stream>>>(
        hb, w2T, (int)MS, Dm, DFF, 0, 0, 0, 0, b2, x, 1.0f, y, nullptr, S);

    // ---- LN2 -> output
    layernorm_kernel<0><<<dim3((unsigned)MS), blk, 0, stream>>>(y, ln2w, ln2b, out, nullptr, Dm);
}
